// GATv2Model_11407433138393
// MI455X (gfx1250) — matrix operand order repaired and checked by reading, not compiled, not run on hardware
//
#include <hip/hip_runtime.h>

#define NN     25000
#define NP     25008          // rows padded to multiple of 16
#define EE     400000
#define EPRIME (EE + NN)      // edges incl. self loops
#define F_IN   133
#define KPAD   160            // 133 padded to 5 * 32
#define HEADS  8
#define CDIM   64
#define HC     512
#define NEG_SLOPE 0.2f
#define BN_EPS    1e-5f

typedef __bf16 bf16_t;
typedef __attribute__((ext_vector_type(8)))  __bf16 v8bf;
typedef __attribute__((ext_vector_type(16))) __bf16 v16bf;
typedef __attribute__((ext_vector_type(8)))  float  v8f;
typedef int v4i_t __attribute__((vector_size(16)));   // matches builtin param type

#if defined(__has_builtin)
#if __has_builtin(__builtin_amdgcn_global_load_async_to_lds_b128) && \
    __has_builtin(__builtin_amdgcn_s_wait_asynccnt)
#define USE_ASYNC_LDS 1
#endif
#endif

// ---------- monotonic float <-> uint encoding for atomicMax on f32 ----------
__device__ __forceinline__ unsigned fenc(float f) {
  unsigned u = __float_as_uint(f);
  return (u & 0x80000000u) ? ~u : (u | 0x80000000u);
}
__device__ __forceinline__ float fdec(unsigned u) {
  u = (u & 0x80000000u) ? (u ^ 0x80000000u) : ~u;
  return __uint_as_float(u);
}

// ---------------------------------------------------------------------------
// Pack x [NN,133] f32 -> Xp [NP,160] bf16, zero padded (rows and K).
// ---------------------------------------------------------------------------
__global__ void pack_x_kernel(const float* __restrict__ x, __bf16* __restrict__ Xp) {
  long long idx = (long long)blockIdx.x * blockDim.x + threadIdx.x;
  if (idx >= (long long)NP * KPAD) return;
  int k   = (int)(idx % KPAD);
  int row = (int)(idx / KPAD);
  float v = (row < NN && k < F_IN) ? x[(size_t)row * F_IN + k] : 0.0f;
  Xp[idx] = (bf16_t)v;
}

// ---------------------------------------------------------------------------
// Pack W [133,512] f32 -> Wp [5][2][512][16] bf16 in B-fragment register order:
//   Wp[((kb*2+hi)*512+col)*16 + j] = W[(kb*32 + hi*16 + j)*512 + col]  (0 if k>=133)
// ---------------------------------------------------------------------------
__global__ void pack_w_kernel(const float* __restrict__ W, __bf16* __restrict__ Wp) {
  int idx = blockIdx.x * blockDim.x + threadIdx.x;   // 5*2*512 = 5120
  if (idx >= 5 * 2 * 512) return;
  int col = idx & 511;
  int hi  = (idx >> 9) & 1;
  int kb5 = idx >> 10;
  __bf16* dst = Wp + (size_t)idx * 16;
  #pragma unroll
  for (int j = 0; j < 16; ++j) {
    int k = kb5 * 32 + hi * 16 + j;
    float v = (k < F_IN) ? W[(size_t)k * HC + col] : 0.0f;
    dst[j] = (bf16_t)v;
  }
}

// ---------------------------------------------------------------------------
// GEMM: Y[NP,512] = Xp[NP,160] @ W[160,512]  (bf16 WMMA, f32 accum)
// Block = 256 threads (8 waves). Block handles one 16-row tile; wave w does
// column tile tn = blockIdx.y*8 + w. A tile staged via LDS (async if available).
// ---------------------------------------------------------------------------
__global__ void __launch_bounds__(256)
gemm_xw_wmma(const __bf16* __restrict__ Xp,
             const __bf16* __restrict__ Wp,
             float* __restrict__ Y) {
  __shared__ __align__(16) __bf16 As[16 * KPAD];   // 5120 B
  const int tid  = threadIdx.x;
  const int wave = tid >> 5;
  const int lane = tid & 31;
  const int hi   = lane >> 4;
  const int l15  = lane & 15;
  const int tm   = blockIdx.x;
  const int tn   = blockIdx.y * 8 + wave;

  // ---- stage A tile: 16 rows x 160 halves = 320 x 16B chunks ----
  const __bf16* gA = Xp + (size_t)tm * 16 * KPAD;
#if USE_ASYNC_LDS
  for (int idx = tid; idx < 320; idx += 256) {
    __builtin_amdgcn_global_load_async_to_lds_b128(
        (__attribute__((address_space(1))) v4i_t*)(gA + idx * 8),
        (__attribute__((address_space(3))) v4i_t*)(As + idx * 8),
        0, 0);
  }
  __builtin_amdgcn_s_wait_asynccnt(0);
#else
  for (int idx = tid; idx < 320; idx += 256) {
    ((v8bf*)As)[idx] = ((const v8bf*)gA)[idx];
  }
#endif
  __syncthreads();

  // ---- 5 K-steps of WMMA ----
  v8f acc = {};
  #pragma unroll
  for (int kb5 = 0; kb5 < 5; ++kb5) {
    // A: lane holds row M=l15; K = kb+16*hi..+15 contiguous
    const __bf16* ap = As + l15 * KPAD + kb5 * 32 + hi * 16;
    v8bf a_lo = *(const v8bf*)(ap);
    v8bf a_hi = *(const v8bf*)(ap + 8);
    v16bf a = __builtin_shufflevector(a_lo, a_hi,
        0, 1, 2, 3, 4, 5, 6, 7, 8, 9, 10, 11, 12, 13, 14, 15);
    // B: contiguous 32B run in packed layout
    const __bf16* bp = Wp + ((size_t)((kb5 * 2 + hi) * 512 + tn * 16 + l15)) * 16;
    v8bf b_lo = *(const v8bf*)(bp);
    v8bf b_hi = *(const v8bf*)(bp + 8);
    v16bf b = __builtin_shufflevector(b_lo, b_hi,
        0, 1, 2, 3, 4, 5, 6, 7, 8, 9, 10, 11, 12, 13, 14, 15);
    acc = __builtin_amdgcn_wmma_f32_16x16x32_bf16(
        false, a, false, b, (short)0, acc, false, false);
  }

  // C/D layout: lane 0-15 -> M=r, lane 16-31 -> M=r+8 ; N = lane&15
  float* yp = Y + (size_t)(tm * 16 + hi * 8) * HC + tn * 16 + l15;
  #pragma unroll
  for (int r = 0; r < 8; ++r) yp[(size_t)r * HC] = acc[r];
}

// ---------------------------------------------------------------------------
// Phase A: per (edge, head) attention logit + segment max via encoded atomicMax
// ---------------------------------------------------------------------------
__global__ void edge_logits_kernel(const int* __restrict__ ei,
                                   const float* __restrict__ xl,
                                   const float* __restrict__ xr,
                                   const float* __restrict__ att,
                                   float* __restrict__ e,
                                   unsigned int* __restrict__ emax) {
  long long idx = (long long)blockIdx.x * blockDim.x + threadIdx.x;
  if (idx >= (long long)EPRIME * HEADS) return;
  int h    = (int)(idx & (HEADS - 1));
  int edge = (int)(idx >> 3);
  int s, d;
  if (edge < EE) { s = ei[edge]; d = ei[EE + edge]; } else { s = d = edge - EE; }

  const float4* pl = (const float4*)(xl + (size_t)s * HC + h * CDIM);
  const float4* pr = (const float4*)(xr + (size_t)d * HC + h * CDIM);
  const float4* pa = (const float4*)(att + h * CDIM);
  float acc = 0.0f;
  #pragma unroll 4
  for (int c4 = 0; c4 < CDIM / 4; ++c4) {
    float4 a = pl[c4], b = pr[c4], w = pa[c4];
    float v0 = a.x + b.x; v0 = v0 > 0.0f ? v0 : NEG_SLOPE * v0;
    float v1 = a.y + b.y; v1 = v1 > 0.0f ? v1 : NEG_SLOPE * v1;
    float v2 = a.z + b.z; v2 = v2 > 0.0f ? v2 : NEG_SLOPE * v2;
    float v3 = a.w + b.w; v3 = v3 > 0.0f ? v3 : NEG_SLOPE * v3;
    acc += v0 * w.x + v1 * w.y + v2 * w.z + v3 * w.w;
  }
  e[idx] = acc;
  atomicMax(&emax[(size_t)d * HEADS + h], fenc(acc));
}

// ---------------------------------------------------------------------------
// Phase B: ee = exp(e - max); scatter-add denominator and weighted messages
// ---------------------------------------------------------------------------
__global__ void edge_aggregate_kernel(const int* __restrict__ ei,
                                      const float* __restrict__ xl,
                                      const float* __restrict__ e,
                                      const unsigned int* __restrict__ emax,
                                      float* __restrict__ den,
                                      float* __restrict__ agg) {
  long long idx = (long long)blockIdx.x * blockDim.x + threadIdx.x;
  if (idx >= (long long)EPRIME * HEADS) return;
  int h    = (int)(idx & (HEADS - 1));
  int edge = (int)(idx >> 3);
  int s, d;
  if (edge < EE) { s = ei[edge]; d = ei[EE + edge]; } else { s = d = edge - EE; }

  float m  = fdec(emax[(size_t)d * HEADS + h]);
  float ee = __expf(e[idx] - m);
  atomicAdd(&den[(size_t)d * HEADS + h], ee);

  const float* msg = xl + (size_t)s * HC + h * CDIM;
  float*       ag  = agg + (size_t)d * HC + h * CDIM;
  #pragma unroll 8
  for (int c = 0; c < CDIM; ++c) atomicAdd(&ag[c], ee * msg[c]);
}

// ---------------------------------------------------------------------------
// GAT output normalize + bias + BatchNorm(eval) + PReLU
// ---------------------------------------------------------------------------
__global__ void finalize_h_kernel(const float* __restrict__ agg,
                                  const float* __restrict__ den,
                                  const float* __restrict__ b1,
                                  const float* __restrict__ gamma,
                                  const float* __restrict__ beta,
                                  const float* __restrict__ mean,
                                  const float* __restrict__ var,
                                  const float* __restrict__ prelu_w,
                                  float* __restrict__ hout) {
  long long idx = (long long)blockIdx.x * blockDim.x + threadIdx.x;
  if (idx >= (long long)NN * HC) return;
  int f    = (int)(idx & (HC - 1));
  int node = (int)(idx >> 9);
  int h    = f >> 6;
  float v = agg[idx] / den[(size_t)node * HEADS + h] + b1[f];
  v = (v - mean[f]) * rsqrtf(var[f] + BN_EPS) * gamma[f] + beta[f];
  float pw = prelu_w[0];
  hout[idx] = (v >= 0.0f) ? v : pw * v;
}

// ---------------------------------------------------------------------------
// GCN: degree, rsqrt norm, per-node 512->1 dot, init with self loop, edge scatter
// ---------------------------------------------------------------------------
__global__ void deg_kernel(const int* __restrict__ ei, float* __restrict__ deg) {
  int idx = blockIdx.x * blockDim.x + threadIdx.x;
  if (idx < EE) atomicAdd(&deg[ei[EE + idx]], 1.0f);
}

__global__ void dinv_kernel(const float* __restrict__ deg, float* __restrict__ dinv) {
  int i = blockIdx.x * blockDim.x + threadIdx.x;
  if (i < NN) dinv[i] = rsqrtf(deg[i] + 1.0f);  // +1 self loop; always >= 1
}

__global__ void gcn_node_kernel(const float* __restrict__ hin,
                                const float* __restrict__ Wg,
                                const float* __restrict__ bg,
                                const float* __restrict__ dinv,
                                float* __restrict__ hg,
                                float* __restrict__ out) {
  int i = blockIdx.x * blockDim.x + threadIdx.x;
  if (i >= NN) return;
  const float4* ph = (const float4*)(hin + (size_t)i * HC);
  const float4* pw = (const float4*)Wg;
  float acc = 0.0f;
  #pragma unroll 8
  for (int j = 0; j < HC / 4; ++j) {
    float4 a = ph[j], w = pw[j];
    acc += a.x * w.x + a.y * w.y + a.z * w.z + a.w * w.w;
  }
  hg[i] = acc;
  float di = dinv[i];
  out[i] = bg[0] + di * di * acc;     // bias + self-loop term; also inits d_out
}

__global__ void gcn_edge_kernel(const int* __restrict__ ei,
                                const float* __restrict__ hg,
                                const float* __restrict__ dinv,
                                float* __restrict__ out) {
  int idx = blockIdx.x * blockDim.x + threadIdx.x;
  if (idx >= EE) return;
  int s = ei[idx], d = ei[EE + idx];
  atomicAdd(&out[d], dinv[s] * dinv[d] * hg[s]);
}

// ---------------------------------------------------------------------------
extern "C" void kernel_launch(void* const* d_in, const int* in_sizes, int n_in,
                              void* d_out, int out_size, void* d_ws, size_t ws_size,
                              hipStream_t stream) {
  const float* x        = (const float*)d_in[0];
  const int*   ei       = (const int*)  d_in[1];
  const float* Wl       = (const float*)d_in[2];
  const float* Wr       = (const float*)d_in[3];
  const float* att      = (const float*)d_in[4];
  const float* b1       = (const float*)d_in[5];
  const float* bn_gamma = (const float*)d_in[6];
  const float* bn_beta  = (const float*)d_in[7];
  const float* bn_mean  = (const float*)d_in[8];
  const float* bn_var   = (const float*)d_in[9];
  const float* prelu_w  = (const float*)d_in[10];
  const float* Wg       = (const float*)d_in[11];
  const float* bg       = (const float*)d_in[12];
  float* out = (float*)d_out;

  // ---- workspace carve-up (deliberate aliasing; launches are stream-ordered) ----
  float*        xl   = (float*)d_ws;                          // [NP*HC]
  float*        xr   = xl + (size_t)NP * HC;                  // [NP*HC]
  float*        agg  = xr;                                    // alias: xr dead after phase A
  float*        hbuf = xl;                                    // alias: xl dead after phase B
  float*        e    = xr + (size_t)NP * HC;                  // [EPRIME*HEADS]
  unsigned int* emax = (unsigned int*)(e + (size_t)EPRIME * HEADS); // [NN*HEADS]
  float*        den  = (float*)emax + (size_t)NN * HEADS;     // [NN*HEADS]
  float*        hg   = den + (size_t)NN * HEADS;              // [NN]
  float*        deg  = hg + NN;                               // [NN]
  float*        dinv = deg + NN;                              // [NN]
  __bf16*       Xp   = (__bf16*)(dinv + NN);                  // [NP*KPAD] bf16
  __bf16*       Wpl  = Xp + (size_t)NP * KPAD;                // [5*2*512*16] bf16
  __bf16*       Wpr  = Wpl + 5 * 2 * 512 * 16;                // [5*2*512*16] bf16

  hipError_t err;
  (void)err;

  // ---- pack to bf16 (padded / fragment-ordered) ----
  long long npk = (long long)NP * KPAD;
  pack_x_kernel<<<(unsigned)((npk + 255) / 256), 256, 0, stream>>>(x, Xp);
  pack_w_kernel<<<(5 * 2 * 512 + 255) / 256, 256, 0, stream>>>(Wl, Wpl);
  pack_w_kernel<<<(5 * 2 * 512 + 255) / 256, 256, 0, stream>>>(Wr, Wpr);

  // ---- GATv2 dense transforms (WMMA bf16) ----
  dim3 gblock(256);
  dim3 ggrid(NP / 16, HC / (16 * 8));
  gemm_xw_wmma<<<ggrid, gblock, 0, stream>>>(Xp, Wpl, xl);
  gemm_xw_wmma<<<ggrid, gblock, 0, stream>>>(Xp, Wpr, xr);

  // ---- edge attention: segment max then exp/sum/aggregate ----
  long long eh = (long long)EPRIME * HEADS;
  unsigned int ehb = (unsigned int)((eh + 255) / 256);
  err = hipMemsetAsync(emax, 0, (size_t)NN * HEADS * sizeof(unsigned int), stream);
  edge_logits_kernel<<<ehb, 256, 0, stream>>>(ei, xl, xr, att, e, emax);
  err = hipMemsetAsync(den, 0, (size_t)NN * HEADS * sizeof(float), stream);
  err = hipMemsetAsync(agg, 0, (size_t)NN * HC * sizeof(float), stream);   // after phase A (aliases xr)
  edge_aggregate_kernel<<<ehb, 256, 0, stream>>>(ei, xl, e, emax, den, agg);

  // ---- normalize + BN + PReLU ----
  unsigned int nhb = (unsigned int)(((long long)NN * HC + 255) / 256);
  finalize_h_kernel<<<nhb, 256, 0, stream>>>(agg, den, b1, bn_gamma, bn_beta,
                                             bn_mean, bn_var, prelu_w, hbuf);

  // ---- GCN ----
  err = hipMemsetAsync(deg, 0, (size_t)NN * sizeof(float), stream);
  deg_kernel<<<(EE + 255) / 256, 256, 0, stream>>>(ei, deg);
  dinv_kernel<<<(NN + 255) / 256, 256, 0, stream>>>(deg, dinv);
  gcn_node_kernel<<<(NN + 255) / 256, 256, 0, stream>>>(hbuf, Wg, bg, dinv, hg, out);
  gcn_edge_kernel<<<(EE + 255) / 256, 256, 0, stream>>>(ei, hg, dinv, out);
}
